// AttentionLayer_43404939493529
// MI455X (gfx1250) — matrix operand order repaired and checked by reading, not compiled, not run on hardware
//
#include <hip/hip_runtime.h>
#include <hip/hip_bf16.h>

typedef __bf16 bf16_t;
typedef __attribute__((ext_vector_type(16))) __bf16 v16bf;
typedef __attribute__((ext_vector_type(8)))  __bf16 v8bf;
typedef __attribute__((ext_vector_type(8)))  float  v8f;
typedef int v4i __attribute__((vector_size(4 * sizeof(int))));

#define S_LEN 2048
#define FDIM  512
#define NROWS (8 * S_LEN)            // B*S = 16384

// ---------------- CDNA5 async global->LDS copy (ASYNCcnt) -------------------
#if defined(__has_builtin)
#  if __has_builtin(__builtin_amdgcn_global_load_async_to_lds_b128)
#    define HAVE_ASYNC_LDS 1
#  endif
#endif
#ifndef HAVE_ASYNC_LDS
#  define HAVE_ASYNC_LDS 0
#endif

static __device__ __forceinline__ void cp16_async(const bf16_t* g, bf16_t* l) {
#if HAVE_ASYNC_LDS
  __builtin_amdgcn_global_load_async_to_lds_b128(
      (__attribute__((address_space(1))) v4i*)(void*)const_cast<bf16_t*>(g),
      (__attribute__((address_space(3))) v4i*)(void*)l, 0, 0);
#else
  *(uint4*)l = *(const uint4*)g;
#endif
}

static __device__ __forceinline__ void wait_async_all() {
#if defined(__has_builtin) && __has_builtin(__builtin_amdgcn_s_wait_asynccnt)
  __builtin_amdgcn_s_wait_asynccnt(0);
#else
  asm volatile("s_wait_asynccnt 0" ::: "memory");
#endif
}

// Assemble a v16bf WMMA fragment from two 16-byte reads.
static __device__ __forceinline__ v16bf ldfrag(const bf16_t* p0, const bf16_t* p1) {
  v8bf lo = *(const v8bf*)p0;
  v8bf hi = *(const v8bf*)p1;
  return __builtin_shufflevector(lo, hi, 0,1,2,3,4,5,6,7,8,9,10,11,12,13,14,15);
}

// ---------------------------------------------------------------------------
// Kernel 0: x + sinusoidal PE -> bf16   (one pass; removes 3x recompute)
// ---------------------------------------------------------------------------
__global__ __launch_bounds__(256) void pe_convert_x(
    const float* __restrict__ x, bf16_t* __restrict__ xbf)
{
  size_t idx = ((size_t)blockIdx.x * 256 + threadIdx.x) * 8;
  int col = (int)(idx & (FDIM - 1));
  int row = (int)(idx >> 9);
  int s   = row & (S_LEN - 1);
  float4 a = *(const float4*)(x + idx);
  float4 b = *(const float4*)(x + idx + 4);
  float xr[8] = {a.x, a.y, a.z, a.w, b.x, b.y, b.z, b.w};
  v8bf pk;
  #pragma unroll
  for (int i = 0; i < 8; ++i) {
    int k     = col + i;
    float e   = (float)(k & ~1) * (1.0f / (float)FDIM);
    float inv = __expf(-e * 9.210340371976184f);     // 10000^-e
    float ang = (float)s * inv;
    float pe  = (k & 1) ? __cosf(ang) : __sinf(ang);
    pk[i] = (bf16_t)(xr[i] + pe);
  }
  *(v8bf*)(xbf + idx) = pk;
}

// ---------------------------------------------------------------------------
// Kernel 1: weight transpose+convert  W[k][n] f32 -> Wt[n][k] bf16
// ---------------------------------------------------------------------------
__global__ __launch_bounds__(256) void w_transpose(
    const float* __restrict__ Wq, const float* __restrict__ Wk,
    const float* __restrict__ Wv,
    bf16_t* __restrict__ Tq, bf16_t* __restrict__ Tk, bf16_t* __restrict__ Tv)
{
  __shared__ float tile[32][33];
  const int z = blockIdx.z;
  const float* W = (z == 0) ? Wq : (z == 1) ? Wk : Wv;
  bf16_t*      O = (z == 0) ? Tq : (z == 1) ? Tk : Tv;
  const int nb = blockIdx.x * 32, kb = blockIdx.y * 32;
  const int tx = threadIdx.x, ty = threadIdx.y;
  #pragma unroll
  for (int j = 0; j < 4; ++j)
    tile[ty + j * 8][tx] = W[(size_t)(kb + ty + j * 8) * FDIM + nb + tx];
  __syncthreads();
  #pragma unroll
  for (int j = 0; j < 4; ++j)
    O[(size_t)(nb + ty + j * 8) * FDIM + kb + tx] = (bf16_t)tile[tx][ty + j * 8];
}

// ---------------------------------------------------------------------------
// Kernel 2: QKV projection GEMM, pure bf16, async double-buffered LDS staging.
//   M=16384, K=512, N=512; grid.z selects Q/K/V. 1/sqrt(512) folded into Q.
// ---------------------------------------------------------------------------
__global__ __launch_bounds__(256) void qkv_gemm(
    const bf16_t* __restrict__ xbf,
    const bf16_t* __restrict__ Tq, const bf16_t* __restrict__ Tk,
    const bf16_t* __restrict__ Tv,
    bf16_t* __restrict__ qo, bf16_t* __restrict__ ko, bf16_t* __restrict__ vo)
{
  __shared__ bf16_t Al[2][128 * 32];   // x tile  [m][k]   8KB each
  __shared__ bf16_t Bl[2][128 * 32];   // Wt tile [n][k]   8KB each

  const int tid  = threadIdx.x;
  const int lane = tid & 31, wid = tid >> 5;
  const int half = lane >> 4, lx = lane & 15;
  const int wm = wid >> 1, wn = wid & 1;          // 4x2 waves, 32x64 each
  const int mbase = blockIdx.y * 128;
  const int nbase = blockIdx.x * 128;
  const int z = blockIdx.z;
  const bf16_t* Wt   = (z == 0) ? Tq : (z == 1) ? Tk : Tv;
  bf16_t*       outp = (z == 0) ? qo : (z == 1) ? ko : vo;

  v8f acc[2][4];
  #pragma unroll
  for (int mt = 0; mt < 2; ++mt)
    #pragma unroll
    for (int nt = 0; nt < 4; ++nt)
      #pragma unroll
      for (int i = 0; i < 8; ++i) acc[mt][nt][i] = 0.f;

  // stage k-step `ks` into buffer `buf`
  auto stage = [&](int ks, int buf) {
    int k0 = ks * 32;
    #pragma unroll
    for (int c = 0; c < 2; ++c) {
      int id  = c * 256 + tid;
      int row = id >> 2;
      int u8  = (id & 3) * 8;
      cp16_async(xbf + (size_t)(mbase + row) * FDIM + k0 + u8, &Al[buf][row * 32 + u8]);
      cp16_async(Wt  + (size_t)(nbase + row) * FDIM + k0 + u8, &Bl[buf][row * 32 + u8]);
    }
  };

  stage(0, 0);
  wait_async_all();
  __syncthreads();

  for (int ks = 0; ks < 16; ++ks) {
    const int cur = ks & 1, nxt = cur ^ 1;
    if (ks + 1 < 16) stage(ks + 1, nxt);

    const int kb = half * 16;                  // A layout: K = kb..kb+15
    v16bf af[2];
    #pragma unroll
    for (int mt = 0; mt < 2; ++mt) {
      int row = wm * 32 + mt * 16 + lx;
      af[mt] = ldfrag(&Al[cur][row * 32 + kb], &Al[cur][row * 32 + kb + 8]);
    }
    #pragma unroll
    for (int nt = 0; nt < 4; ++nt) {
      int n = wn * 64 + nt * 16 + lx;          // B layout: col=lane&15, K=16*half..+15
      v16bf bfb = ldfrag(&Bl[cur][n * 32 + half * 16], &Bl[cur][n * 32 + half * 16 + 8]);
      #pragma unroll
      for (int mt = 0; mt < 2; ++mt)
        acc[mt][nt] = __builtin_amdgcn_wmma_f32_16x16x32_bf16(
            false, af[mt], false, bfb, (short)0, acc[mt][nt], false, false);
    }
    wait_async_all();
    __syncthreads();
  }

  const float scale = (z == 0) ? 0.04419417382415922f : 1.0f;
  #pragma unroll
  for (int mt = 0; mt < 2; ++mt)
    #pragma unroll
    for (int nt = 0; nt < 4; ++nt)
      #pragma unroll
      for (int i = 0; i < 8; ++i) {
        int gm = mbase + wm * 32 + mt * 16 + i + 8 * half;
        int gn = nbase + wn * 64 + nt * 16 + lx;
        outp[(size_t)gm * FDIM + gn] = (bf16_t)(acc[mt][nt][i] * scale);
      }
}

// ---------------------------------------------------------------------------
// Kernel 3: flash attention, online softmax, bf16 WMMA, fp32 accum.
//   64 queries/block, 8 waves = 4 query-groups x 2 D-halves.
//   32-key blocks, double-buffered: K via async copies, V^T via lane scatter.
// ---------------------------------------------------------------------------
__global__ __launch_bounds__(256, 1) void flash_attn(
    const bf16_t* __restrict__ Qg, const bf16_t* __restrict__ Kg,
    const bf16_t* __restrict__ Vg, float* __restrict__ out)
{
  __shared__ bf16_t Qs [64 * 512];        // 64KB  [q][d]
  __shared__ bf16_t Ks [2][32 * 512];     // 64KB  [key][d]
  __shared__ bf16_t VTs[2][512 * 32];     // 64KB  [d][key]
  __shared__ bf16_t Ps [8][16 * 32];      //  8KB  per-wave P scratch

  const int tid  = threadIdx.x;
  const int lane = tid & 31, wid = tid >> 5;
  const int half = lane >> 4, lx = lane & 15;
  const int mg = wid >> 1;                // query group (16 rows)
  const int dh = wid & 1;                 // output-dim half (256 dims)
  const int qbase = blockIdx.x * 64;
  const int b     = blockIdx.y;

  // Stage Q tile via async copies (once).
  #pragma unroll 4
  for (int pass = 0; pass < 16; ++pass) {
    int row = (tid >> 6) + pass * 4;
    int u   = (tid & 63);
    cp16_async(Qg + ((size_t)b * S_LEN + qbase + row) * FDIM + u * 8,
               &Qs[row * 512 + u * 8]);
  }

  auto stage_k = [&](int blk, int buf) {     // async K block (32 x 512)
    #pragma unroll 4
    for (int pass = 0; pass < 8; ++pass) {
      int row = (tid >> 6) + pass * 4;
      int u   = (tid & 63);
      cp16_async(Kg + ((size_t)b * S_LEN + blk * 32 + row) * FDIM + u * 8,
                 &Ks[buf][row * 512 + u * 8]);
    }
  };
  auto stage_v = [&](int blk, int buf) {     // V block transposed via lanes
    #pragma unroll 4
    for (int pass = 0; pass < 8; ++pass) {
      int row = (tid >> 6) + pass * 4;
      int u   = (tid & 63);
      uint4 vv = *(const uint4*)(Vg + ((size_t)b * S_LEN + blk * 32 + row) * FDIM + u * 8);
      const bf16_t* pv = (const bf16_t*)&vv;
      #pragma unroll
      for (int i = 0; i < 8; ++i) VTs[buf][(u * 8 + i) * 32 + row] = pv[i];
    }
  };

  float m_i[8], l_i[8];
  v8f O[16];
  #pragma unroll
  for (int i = 0; i < 8; ++i) { m_i[i] = -1e30f; l_i[i] = 0.f; }
  #pragma unroll
  for (int nt = 0; nt < 16; ++nt)
    #pragma unroll
    for (int i = 0; i < 8; ++i) O[nt][i] = 0.f;

  stage_k(0, 0);
  stage_v(0, 0);
  wait_async_all();
  __syncthreads();

  const int NBLK = S_LEN / 32;               // 64 key blocks
  for (int blk = 0; blk < NBLK; ++blk) {
    const int cur = blk & 1, nxt = cur ^ 1;
    if (blk + 1 < NBLK) { stage_k(blk + 1, nxt); stage_v(blk + 1, nxt); }

    // --- scores = Q.K^T (16 queries x 32 keys), full 512-dim reduction ---
    v8f sc[2];
    #pragma unroll
    for (int nt = 0; nt < 2; ++nt)
      #pragma unroll
      for (int i = 0; i < 8; ++i) sc[nt][i] = 0.f;

    #pragma unroll
    for (int kf = 0; kf < 16; ++kf) {
      int row = mg * 16 + lx;
      int kb  = kf * 32 + half * 16;
      v16bf aq = ldfrag(&Qs[row * 512 + kb], &Qs[row * 512 + kb + 8]);
      #pragma unroll
      for (int nt = 0; nt < 2; ++nt) {
        int key = nt * 16 + lx;
        int kd  = kf * 32 + half * 16;
        v16bf bk = ldfrag(&Ks[cur][key * 512 + kd], &Ks[cur][key * 512 + kd + 8]);
        sc[nt] = __builtin_amdgcn_wmma_f32_16x16x32_bf16(
            false, aq, false, bk, (short)0, sc[nt], false, false);
      }
    }

    // --- online softmax (1/sqrt(d) pre-folded into Q) ---
    float alpha[8];
    #pragma unroll
    for (int i = 0; i < 8; ++i) {
      float mx = fmaxf(sc[0][i], sc[1][i]);
      mx = fmaxf(mx, __shfl_xor(mx, 1));
      mx = fmaxf(mx, __shfl_xor(mx, 2));
      mx = fmaxf(mx, __shfl_xor(mx, 4));
      mx = fmaxf(mx, __shfl_xor(mx, 8));
      float mn = fmaxf(m_i[i], mx);
      alpha[i] = __expf(m_i[i] - mn);
      m_i[i] = mn;
      float rs = 0.f;
      #pragma unroll
      for (int nt = 0; nt < 2; ++nt) {
        float p = __expf(sc[nt][i] - mn);
        sc[nt][i] = p;
        rs += p;
      }
      rs += __shfl_xor(rs, 1);
      rs += __shfl_xor(rs, 2);
      rs += __shfl_xor(rs, 4);
      rs += __shfl_xor(rs, 8);
      l_i[i] = l_i[i] * alpha[i] + rs;
    }
    #pragma unroll
    for (int nt = 0; nt < 16; ++nt)
      #pragma unroll
      for (int i = 0; i < 8; ++i) O[nt][i] *= alpha[i];

    // --- P (C layout) -> bf16 wave-private scratch, then A-layout reload ---
    #pragma unroll
    for (int nt = 0; nt < 2; ++nt)
      #pragma unroll
      for (int i = 0; i < 8; ++i)
        Ps[wid][(i + 8 * half) * 32 + nt * 16 + lx] = (bf16_t)sc[nt][i];
    asm volatile("s_wait_dscnt 0" ::: "memory");   // intra-wave DS ordering

    // --- O += P.V for this wave's 256 output dims (K = 32 keys = 1 wmma) ---
    {
      int kb = half * 16;
      v16bf ap = ldfrag(&Ps[wid][lx * 32 + kb], &Ps[wid][lx * 32 + kb + 8]);
      #pragma unroll
      for (int nt = 0; nt < 16; ++nt) {
        int dcol = dh * 256 + nt * 16 + lx;
        v16bf bv = ldfrag(&VTs[cur][dcol * 32 + half * 16],
                          &VTs[cur][dcol * 32 + half * 16 + 8]);
        O[nt] = __builtin_amdgcn_wmma_f32_16x16x32_bf16(
            false, ap, false, bv, (short)0, O[nt], false, false);
      }
    }

    wait_async_all();       // next block's async K (and Q on blk 0) landed
    __syncthreads();        // everyone done with `cur`, scatter of `nxt` visible
  }

  #pragma unroll
  for (int i = 0; i < 8; ++i) l_i[i] = 1.0f / l_i[i];
  #pragma unroll
  for (int nt = 0; nt < 16; ++nt)
    #pragma unroll
    for (int i = 0; i < 8; ++i) {
      int q = qbase + mg * 16 + i + 8 * half;
      int d = dh * 256 + nt * 16 + lx;
      out[((size_t)b * S_LEN + q) * FDIM + d] = O[nt][i] * l_i[i];
    }
}

// ---------------------------------------------------------------------------
extern "C" void kernel_launch(void* const* d_in, const int* in_sizes, int n_in,
                              void* d_out, int out_size, void* d_ws, size_t ws_size,
                              hipStream_t stream) {
  const float* x  = (const float*)d_in[0];
  const float* Wq = (const float*)d_in[1];
  const float* Wk = (const float*)d_in[2];
  const float* Wv = (const float*)d_in[3];
  float* out = (float*)d_out;

  const size_t NE = (size_t)NROWS * FDIM;    // 16384*512
  const size_t WE = (size_t)FDIM * FDIM;     // 512*512
  bf16_t* xbf = (bf16_t*)d_ws;
  bf16_t* wtq = xbf + NE;
  bf16_t* wtk = wtq + WE;
  bf16_t* wtv = wtk + WE;
  bf16_t* qbf = wtv + WE;
  bf16_t* kbf = qbf + NE;
  bf16_t* vbf = kbf + NE;

  pe_convert_x<<<dim3((unsigned)(NE / 8 / 256)), 256, 0, stream>>>(x, xbf);
  w_transpose<<<dim3(16, 16, 3), dim3(32, 8), 0, stream>>>(Wq, Wk, Wv, wtq, wtk, wtv);
  qkv_gemm<<<dim3(4, 128, 3), 256, 0, stream>>>(xbf, wtq, wtk, wtv, qbf, kbf, vbf);
  flash_attn<<<dim3(32, 8), 256, 0, stream>>>(qbf, kbf, vbf, out);
}
